// PerforantPR_68547678044339
// MI455X (gfx1250) — matrix operand order repaired and checked by reading, not compiled, not run on hardware
//
#include <hip/hip_runtime.h>
#include <hip/hip_bf16.h>

typedef __attribute__((ext_vector_type(16))) __bf16 v16bf;
typedef __attribute__((ext_vector_type(8)))  float  v8f;

enum { BB = 1024, DIN = 4096, DHID = 4096, DOUT = 2048 };
#define LDS_STRIDE 34   // ushort stride; 17 dwords -> conflict-free for 16/128-row access

__device__ __forceinline__ unsigned short f32_to_bf16_rne(float f) {
    unsigned int u = __float_as_uint(f);
    unsigned int r = u + 0x7fffu + ((u >> 16) & 1u);
    return (unsigned short)(r >> 16);
}
__device__ __forceinline__ float bf16_bits_to_f32(unsigned short h) {
    return __uint_as_float(((unsigned int)h) << 16);
}

union Frag { v16bf v; unsigned int u[8]; };

// C[M,N] = act(A[M,K] @ W[K,N] + bias), split-bf16 WMMA (hi*hi + hi*lo + lo*hi),
// software-pipelined: next K-tile's global loads overlap current tile's WMMAs.
// A-fragments are loaded one row-tile at a time to keep live VGPRs < 256.
__global__ __launch_bounds__(256)
void gemm_bf16x3_kernel(const float* __restrict__ A, const float* __restrict__ W,
                        const float* __restrict__ bias, float* __restrict__ C,
                        int M, int N, int K, int leaky)
{
    __shared__ unsigned short AsHi[128 * LDS_STRIDE];
    __shared__ unsigned short AsLo[128 * LDS_STRIDE];
    __shared__ unsigned short BsHi[128 * LDS_STRIDE];   // stored transposed: [n][k]
    __shared__ unsigned short BsLo[128 * LDS_STRIDE];

    const int tid  = threadIdx.x;
    const int wave = tid >> 5;
    const int lane = tid & 31;
    const int waveRow = wave >> 2;      // 0..1  (64 rows each)
    const int waveCol = wave & 3;       // 0..3  (32 cols each)
    const int m0 = blockIdx.y * 128;
    const int n0 = blockIdx.x * 128;
    const int half = lane >> 4;
    const int l16  = lane & 15;

    v8f zero = {0.f,0.f,0.f,0.f,0.f,0.f,0.f,0.f};
    v8f acc[4][2];
    #pragma unroll
    for (int i = 0; i < 4; i++)
        #pragma unroll
        for (int j = 0; j < 2; j++) acc[i][j] = zero;

    float aReg[16], bReg[16];

    // ---- tile fetch: A 128x32 row-major, B(W) 32x128 row-major ----
    auto fetchTiles = [&](int kt) {
        #pragma unroll
        for (int e = 0; e < 16; e++) {
            int idx = tid + e * 256;
            aReg[e] = A[(size_t)(m0 + (idx >> 5)) * K + kt + (idx & 31)];
        }
        #pragma unroll
        for (int e = 0; e < 16; e++) {
            int idx = tid + e * 256;
            bReg[e] = W[(size_t)(kt + (idx >> 7)) * N + n0 + (idx & 127)];
        }
    };

    fetchTiles(0);   // prologue

    for (int kt = 0; kt < K; kt += 32) {
        // ---- convert staged registers -> LDS bf16 hi/lo ----
        #pragma unroll
        for (int e = 0; e < 16; e++) {
            int idx = tid + e * 256;
            int r = idx >> 5, c = idx & 31;
            float x = aReg[e];
            unsigned short hi = f32_to_bf16_rne(x);
            AsHi[r * LDS_STRIDE + c] = hi;
            AsLo[r * LDS_STRIDE + c] = f32_to_bf16_rne(x - bf16_bits_to_f32(hi));
        }
        #pragma unroll
        for (int e = 0; e < 16; e++) {
            int idx = tid + e * 256;
            int r = idx >> 7, c = idx & 127;
            float x = bReg[e];
            unsigned short hi = f32_to_bf16_rne(x);
            BsHi[c * LDS_STRIDE + r] = hi;                       // transposed [n][k]
            BsLo[c * LDS_STRIDE + r] = f32_to_bf16_rne(x - bf16_bits_to_f32(hi));
        }
        __syncthreads();

        // ---- kick off next K-tile global loads (hide under WMMAs) ----
        if (kt + 32 < K) {
            fetchTiles(kt + 32);
            if (kt + 64 < K) {   // shallow prefetch two tiles ahead (global_prefetch_b8)
                __builtin_prefetch(&A[(size_t)(m0 + (tid >> 5)) * K + kt + 64 + (tid & 31)], 0, 1);
                __builtin_prefetch(&W[(size_t)(kt + 64 + (tid >> 7)) * N + n0 + (tid & 127)], 0, 1);
            }
        }

        // ---- B fragments (reused by all 4 A row-tiles) ----
        Frag bHi[2], bLo[2];
        #pragma unroll
        for (int j = 0; j < 2; j++) {
            int col = waveCol * 32 + j * 16 + l16;
            const unsigned int* pHi = (const unsigned int*)&BsHi[col * LDS_STRIDE];
            const unsigned int* pLo = (const unsigned int*)&BsLo[col * LDS_STRIDE];
            #pragma unroll
            for (int v = 0; v < 8; v++) {
                int k = half * 16 + v * 2;
                bHi[j].u[v] = pHi[k >> 1];
                bLo[j].u[v] = pLo[k >> 1];
            }
        }

        // ---- per row-tile: load A fragments then 6 WMMAs (limits live VGPRs) ----
        #pragma unroll
        for (int i = 0; i < 4; i++) {
            Frag aHi, aLo;
            int row = waveRow * 64 + i * 16 + l16;
            const unsigned int* pHi = (const unsigned int*)&AsHi[row * LDS_STRIDE];
            const unsigned int* pLo = (const unsigned int*)&AsLo[row * LDS_STRIDE];
            #pragma unroll
            for (int v = 0; v < 8; v++) {
                int k = half * 16 + v * 2;
                aHi.u[v] = pHi[k >> 1];
                aLo.u[v] = pLo[k >> 1];
            }
            #pragma unroll
            for (int j = 0; j < 2; j++) {
                acc[i][j] = __builtin_amdgcn_wmma_f32_16x16x32_bf16(
                    false, aHi.v, false, bHi[j].v, (short)0, acc[i][j], false, false);
                acc[i][j] = __builtin_amdgcn_wmma_f32_16x16x32_bf16(
                    false, aHi.v, false, bLo[j].v, (short)0, acc[i][j], false, false);
                acc[i][j] = __builtin_amdgcn_wmma_f32_16x16x32_bf16(
                    false, aLo.v, false, bHi[j].v, (short)0, acc[i][j], false, false);
            }
        }
        __syncthreads();
    }

    // ---- epilogue: bias + optional leaky_relu, C/D layout: VGPR v -> M = half*8+v ----
    #pragma unroll
    for (int i = 0; i < 4; i++) {
        #pragma unroll
        for (int j = 0; j < 2; j++) {
            int col = n0 + waveCol * 32 + j * 16 + l16;
            float bv = bias[col];
            #pragma unroll
            for (int v = 0; v < 8; v++) {
                int row = m0 + waveRow * 64 + i * 16 + half * 8 + v;
                float val = acc[i][j][v] + bv;
                if (leaky) val = (val > 0.f) ? val : 0.01f * val;
                C[(size_t)row * N + col] = val;
            }
        }
    }
}

// BCE-with-logits partial sums + in-place sigmoid (y overwrites logits)
__global__ __launch_bounds__(256)
void loss_sigmoid_kernel(float* logits_y, const float* __restrict__ targets,
                         float* __restrict__ partials)
{
    __shared__ float red[256];
    int tid = threadIdx.x;
    float s = 0.f;
    #pragma unroll
    for (int e = 0; e < 8; e++) {
        size_t i = (size_t)blockIdx.x * 2048 + e * 256 + tid;
        float l = logits_y[i];
        float t = targets[i];
        s += fmaxf(l, 0.f) - l * t + log1pf(expf(-fabsf(l)));
        float sig = 1.f / (1.f + expf(-l));
        logits_y[i] = fminf(fmaxf(sig, 0.f), 1.f);
    }
    red[tid] = s;
    __syncthreads();
    for (int st = 128; st > 0; st >>= 1) {
        if (tid < st) red[tid] += red[tid + st];
        __syncthreads();
    }
    if (tid == 0) partials[blockIdx.x] = red[0];
}

__global__ __launch_bounds__(256)
void loss_reduce_kernel(const float* __restrict__ partials, float* __restrict__ out,
                        int n, float invCount)
{
    __shared__ float red[256];
    int tid = threadIdx.x;
    float s = 0.f;
    for (int i = tid; i < n; i += 256) s += partials[i];
    red[tid] = s;
    __syncthreads();
    for (int st = 128; st > 0; st >>= 1) {
        if (tid < st) red[tid] += red[tid + st];
        __syncthreads();
    }
    if (tid == 0) out[0] = red[0] * invCount;
}

// One block per row: top-75 mask -> sum-norm -> softmax -> gain -> top-51 shift
__global__ __launch_bounds__(256)
void row_kernel(const float* __restrict__ y, float* __restrict__ pr,
                float* __restrict__ zin, float* __restrict__ zcue)
{
    __shared__ float sy[DOUT];
    __shared__ float red[256];
    __shared__ int cnt;
    const int tid = threadIdx.x;
    const size_t base = (size_t)blockIdx.x * DOUT;

    #pragma unroll
    for (int e = 0; e < 8; e++) sy[e * 256 + tid] = y[base + e * 256 + tid];
    __syncthreads();

    // bisect for 75th-largest threshold of y (values in (0,1))
    float lo = 0.f, hi = 1.f;
    for (int it = 0; it < 50; it++) {
        float mid = 0.5f * (lo + hi);
        if (tid == 0) cnt = 0;
        __syncthreads();
        int c = 0;
        #pragma unroll
        for (int e = 0; e < 8; e++) c += (sy[e * 256 + tid] > mid) ? 1 : 0;
        atomicAdd(&cnt, c);
        __syncthreads();
        if (cnt >= 75) lo = mid; else hi = mid;
        __syncthreads();
    }
    float t75 = lo;

    // masked sum (deterministic tree reduce)
    float s = 0.f;
    #pragma unroll
    for (int e = 0; e < 8; e++) { float v = sy[e * 256 + tid]; if (v > t75) s += v; }
    red[tid] = s; __syncthreads();
    for (int st = 128; st > 0; st >>= 1) { if (tid < st) red[tid] += red[tid + st]; __syncthreads(); }
    float S = red[0]; __syncthreads();
    float scale = 1.f / S + 1e-13f;   // (1/sum + EPS) * SUM_NORM(=1)

    // y2 = masked & scaled, overwrite sy
    #pragma unroll
    for (int e = 0; e < 8; e++) {
        int i = e * 256 + tid;
        float v = sy[i];
        sy[i] = (v > t75) ? v * scale : 0.f;
    }
    __syncthreads();

    // softmax over full row (zeros included), then gain
    float mx = 0.f;
    #pragma unroll
    for (int e = 0; e < 8; e++) mx = fmaxf(mx, sy[e * 256 + tid]);
    red[tid] = mx; __syncthreads();
    for (int st = 128; st > 0; st >>= 1) { if (tid < st) red[tid] = fmaxf(red[tid], red[tid + st]); __syncthreads(); }
    mx = red[0]; __syncthreads();

    float zs = 0.f;
    #pragma unroll
    for (int e = 0; e < 8; e++) zs += expf(sy[e * 256 + tid] - mx);
    red[tid] = zs; __syncthreads();
    for (int st = 128; st > 0; st >>= 1) { if (tid < st) red[tid] += red[tid + st]; __syncthreads(); }
    float Z = red[0]; __syncthreads();
    float g = 2.0f / Z;               // GAIN = 2
    #pragma unroll
    for (int e = 0; e < 8; e++) { int i = e * 256 + tid; sy[i] = expf(sy[i] - mx) * g; }
    __syncthreads();

    // 51st-largest of pr -> shift = 1 - 2*kth  (since max((1-pr)*mask) = 1 - kth)
    lo = 0.f; hi = 2.f;
    for (int it = 0; it < 50; it++) {
        float mid = 0.5f * (lo + hi);
        if (tid == 0) cnt = 0;
        __syncthreads();
        int c = 0;
        #pragma unroll
        for (int e = 0; e < 8; e++) c += (sy[e * 256 + tid] > mid) ? 1 : 0;
        atomicAdd(&cnt, c);
        __syncthreads();
        if (cnt >= 51) lo = mid; else hi = mid;
        __syncthreads();
    }
    float shift = 1.f - 2.f * (0.5f * (lo + hi));

    #pragma unroll
    for (int e = 0; e < 8; e++) {
        int i = e * 256 + tid;
        float p = sy[i];
        float z = 2.f * p - 1.f;
        pr[base + i]   = p;
        zin[base + i]  = z;
        zcue[base + i] = z + shift;
    }
}

extern "C" void kernel_launch(void* const* d_in, const int* in_sizes, int n_in,
                              void* d_out, int out_size, void* d_ws, size_t ws_size,
                              hipStream_t stream)
{
    const float* inputs  = (const float*)d_in[0];
    const float* targets = (const float*)d_in[1];
    const float* w_enc   = (const float*)d_in[2];
    const float* b_enc   = (const float*)d_in[3];
    const float* w_dec   = (const float*)d_in[4];
    const float* b_dec   = (const float*)d_in[5];

    float* out      = (float*)d_out;
    float* hidden   = (float*)d_ws;                      // BB*DHID f32 (16 MB)
    float* logits   = hidden + (size_t)BB * DHID;        // BB*DOUT f32 (8 MB), becomes y
    float* partials = logits + (size_t)BB * DOUT;        // 1024 f32

    dim3 blk(256);
    // hidden = leaky_relu(inputs @ w_enc + b_enc)
    gemm_bf16x3_kernel<<<dim3(DHID / 128, BB / 128), blk, 0, stream>>>(
        inputs, w_enc, b_enc, hidden, BB, DHID, DIN, 1);
    // logits = hidden @ w_dec + b_dec
    gemm_bf16x3_kernel<<<dim3(DOUT / 128, BB / 128), blk, 0, stream>>>(
        hidden, w_dec, b_dec, logits, BB, DOUT, DHID, 0);

    const int nElems = BB * DOUT;
    const int nBlk = nElems / 2048;   // 1024
    loss_sigmoid_kernel<<<nBlk, blk, 0, stream>>>(logits, targets, partials);
    loss_reduce_kernel<<<1, blk, 0, stream>>>(partials, out, nBlk, 1.f / (float)nElems);

    float* pr   = out + 1;
    float* zin  = pr  + (size_t)nElems;
    float* zcue = zin + (size_t)nElems;
    row_kernel<<<BB, blk, 0, stream>>>(logits, pr, zin, zcue);
}
